// GCNRU_2388001817260
// MI455X (gfx1250) — matrix operand order repaired and checked by reading, not compiled, not run on hardware
//
#include <hip/hip_runtime.h>
#include <hip/hip_bf16.h>

typedef __attribute__((ext_vector_type(16))) __bf16          v16bf;
typedef __attribute__((ext_vector_type(16))) unsigned short  v16u;
typedef __attribute__((ext_vector_type(8)))  unsigned        v8u;
typedef __attribute__((ext_vector_type(8)))  float           v8f;

__device__ __forceinline__ unsigned short f2bf(float f) {
  // f32 -> bf16, round-to-nearest (ties away)
  unsigned u = __builtin_bit_cast(unsigned, f) + 0x8000u;
  return (unsigned short)(u >> 16);
}

__device__ __forceinline__ unsigned f2bf2(float lo, float hi) {
  // two rounded bf16 packed into one dword: 2x v_add + 1x v_perm_b32
  unsigned ul = __builtin_bit_cast(unsigned, lo) + 0x8000u;
  unsigned uh = __builtin_bit_cast(unsigned, hi) + 0x8000u;
  return __builtin_amdgcn_perm(uh, ul, 0x07060302u);
}

// ---------------------------------------------------------------------------
// WMMA GEMM: C[M x NT*16] = A[M x KD] (bf16, lda) @ W[KD x NT*16] (+bias)(+relu)
// A already bf16 in memory; f32 accumulate via v_wmma_f32_16x16x32_bf16.
// One wave -> 16 rows x NT*16 cols. 8 waves / block.
// ---------------------------------------------------------------------------
template<int KD, int NT, int EPI>  // EPI: 0=none, 1=+bias, 2=+bias+relu
__global__ __launch_bounds__(256)
void gemm_wmma(const unsigned short* __restrict__ A, int lda,
               const float* __restrict__ W,
               const float* __restrict__ bias,
               float* __restrict__ C, int ldc, int M) {
  constexpr int NC = NT * 16;
  constexpr int ST = KD + 16;              // LDS col stride (elems); (ST*2)%32==0
  __shared__ unsigned short ldsW[NC * ST]; // W^T in bf16: ldsW[c*ST + k]

  const int tid = threadIdx.x;
  for (int i = tid; i < KD * NC; i += 256) {
    int k = i / NC, c = i - k * NC;
    ldsW[c * ST + k] = f2bf(W[i]);
  }
  __syncthreads();

  const int wave  = tid >> 5;
  const int lane  = tid & 31;
  const int tiles = (M + 15) >> 4;
  const int tile  = blockIdx.x * 8 + wave;
  if (tile >= tiles) return;               // after barrier: safe

  const int row0 = tile << 4;
  const int hlf  = lane >> 4;              // 0: lanes 0-15, 1: lanes 16-31
  const int mcol = lane & 15;

  v8f zero = {};
  v8f acc[NT];
#pragma unroll
  for (int t = 0; t < NT; ++t) acc[t] = zero;

  // A fragment (16x32, contiguous k order, same as B): row = row0 + (lane&15);
  // elements 0..7  -> K = k0 + hlf*16 + j       (16B contiguous)
  // elements 8..15 -> K = k0 + hlf*16 + j       (16B contiguous)
  const unsigned short* arow = A + (size_t)(row0 + mcol) * lda + hlf * 16;
  const int kbB = hlf * 16;                // B: element j -> K = k0 + hlf*16 + j

  for (int k0 = 0; k0 < KD; k0 += 32) {
    uint4 lo = *(const uint4*)(arow + k0);        // global_load_b128
    uint4 hi = *(const uint4*)(arow + k0 + 8);    // global_load_b128
    v8u au;
    au[0] = lo.x; au[1] = lo.y; au[2] = lo.z; au[3] = lo.w;
    au[4] = hi.x; au[5] = hi.y; au[6] = hi.z; au[7] = hi.w;
    v16bf af = __builtin_bit_cast(v16bf, au);
#pragma unroll
    for (int t = 0; t < NT; ++t) {
      int col = t * 16 + mcol;
      v16u bu = *(const v16u*)(&ldsW[col * ST + k0 + kbB]); // 2x ds_load_b128
      v16bf bf = __builtin_bit_cast(v16bf, bu);
      acc[t] = __builtin_amdgcn_wmma_f32_16x16x32_bf16(
          false, af, false, bf, (short)0, acc[t], false, false);
    }
  }

  // D layout: element v -> row = row0 + v + 8*hlf, col = t*16 + (lane&15)
  if (row0 + 16 <= M) {                    // wave-uniform full-tile fast path
#pragma unroll
    for (int t = 0; t < NT; ++t) {
      int col = t * 16 + mcol;
      float bv = 0.0f;
      if constexpr (EPI > 0) bv = bias[col];
      float* cp = C + (size_t)(row0 + hlf * 8) * ldc + col;
#pragma unroll
      for (int v = 0; v < 8; ++v) {
        float xv = acc[t][v] + bv;
        if constexpr (EPI == 2) xv = fmaxf(xv, 0.0f);
        cp[(size_t)v * ldc] = xv;
      }
    }
  } else {                                 // partial last tile (M % 16 != 0)
#pragma unroll
    for (int t = 0; t < NT; ++t) {
      int col = t * 16 + mcol;
      float bv = 0.0f;
      if constexpr (EPI > 0) bv = bias[col];
#pragma unroll
      for (int v = 0; v < 8; ++v) {
        int r = row0 + v + hlf * 8;
        if (r < M) {
          float xv = acc[t][v] + bv;
          if constexpr (EPI == 2) xv = fmaxf(xv, 0.0f);
          C[(size_t)r * ldc + col] = xv;
        }
      }
    }
  }
}

// ---------------------------------------------------------------------------
// Elementwise / graph kernels
// ---------------------------------------------------------------------------
__global__ void k_fill(float* p, float v, int n) {
  int i = blockIdx.x * 256 + threadIdx.x;
  if (i < n) p[i] = v;
}

__global__ void k_zero4(float4* p, int n4) {
  int i = blockIdx.x * 256 + threadIdx.x;
  if (i < n4) p[i] = float4{0.f, 0.f, 0.f, 0.f};
}

// f32 -> bf16 bulk convert (4 elems/thread)
__global__ void k_cvt(const float4* __restrict__ in, uint2* __restrict__ out, int n4) {
  int i = blockIdx.x * 256 + threadIdx.x;
  if (i < n4) {
    float4 v = in[i];
    out[i] = uint2{f2bf2(v.x, v.y), f2bf2(v.z, v.w)};
  }
}

__global__ void k_edge_deg(const int* __restrict__ dst, float* __restrict__ deg, int E) {
  int i = blockIdx.x * 256 + threadIdx.x;
  if (i < E) atomicAdd(&deg[dst[i]], 1.0f);
}

__global__ void k_rsqrt(float* p, int n) {
  int i = blockIdx.x * 256 + threadIdx.x;
  if (i < n) p[i] = rsqrtf(fmaxf(p[i], 1.0f));
}

// one wave32 per edge: acc[dst] += T[src] * dinv[src]*dinv[dst]  (96 floats/row)
__global__ void k_edge_agg(const float* __restrict__ T,
                           const int* __restrict__ src, const int* __restrict__ dst,
                           const float* __restrict__ dinv,
                           float* __restrict__ out, int ldo, int E) {
  int w    = (blockIdx.x * blockDim.x + threadIdx.x) >> 5;
  int lane = threadIdx.x & 31;
  if (w >= E) return;
  int s = src[w], d = dst[w];
  float nrm = dinv[s] * dinv[d];
  const float* trow = T + (size_t)s * 96;
  float* orow = out + (size_t)d * ldo;
#pragma unroll
  for (int i = 0; i < 3; ++i) {
    int c = lane + 32 * i;
    atomicAdd(&orow[c], trow[c] * nrm);   // feature matrix is L2-resident
  }
}

// h = relu(acc + T*dinv^2 + b), emitted directly as bf16 (float4 granularity)
__global__ void k_finish(const float* __restrict__ accb, int lda_,
                         const float* __restrict__ T, const float* __restrict__ dinv,
                         const float* __restrict__ b,
                         unsigned short* __restrict__ outB, int ldo, int N) {
  int idx = blockIdx.x * 256 + threadIdx.x;   // over N*24 float4 chunks
  if (idx >= N * 24) return;
  int i = idx / 24, c = (idx - 24 * i) * 4;
  float di = dinv[i];
  float s = di * di;
  float4 tv = *(const float4*)(T + (size_t)i * 96 + c);
  float4 ov = *(const float4*)(accb + (size_t)i * lda_ + c);
  float4 bv = *(const float4*)(b + c);
  float rx = fmaxf(ov.x + tv.x * s + bv.x, 0.f);
  float ry = fmaxf(ov.y + tv.y * s + bv.y, 0.f);
  float rz = fmaxf(ov.z + tv.z * s + bv.z, 0.f);
  float rw = fmaxf(ov.w + tv.w * s + bv.w, 0.f);
  *(uint2*)(outB + (size_t)i * ldo + c) = uint2{f2bf2(rx, ry), f2bf2(rz, rw)};
}

// one wave32 per node: emb[batch[i]] += hjk[i]
__global__ void k_pool(const float* __restrict__ T, const int* __restrict__ batch,
                       float* __restrict__ emb, int N) {
  int w    = (blockIdx.x * blockDim.x + threadIdx.x) >> 5;
  int lane = threadIdx.x & 31;
  if (w >= N) return;
  int g = batch[w];
  const float* tr = T + (size_t)w * 96;
  float* er = emb + (size_t)g * 96;
#pragma unroll
  for (int i = 0; i < 3; ++i) {
    int c = lane + 32 * i;
    atomicAdd(&er[c], tr[c]);
  }
}

// ---------------------------------------------------------------------------
extern "C" void kernel_launch(void* const* d_in, const int* in_sizes, int n_in,
                              void* d_out, int out_size, void* d_ws, size_t ws_size,
                              hipStream_t stream) {
  const float* x    = (const float*)d_in[0];
  const int*   ei   = (const int*)  d_in[1];
  const int*   bat  = (const int*)  d_in[2];
  const float* W0   = (const float*)d_in[3];  const float* b0  = (const float*)d_in[4];
  const float* W1   = (const float*)d_in[5];  const float* b1  = (const float*)d_in[6];
  const float* W2   = (const float*)d_in[7];  const float* b2  = (const float*)d_in[8];
  const float* Wjk  = (const float*)d_in[9];  const float* bjk = (const float*)d_in[10];
  const float* Wm1  = (const float*)d_in[11]; const float* bm1 = (const float*)d_in[12];
  const float* Wm2  = (const float*)d_in[13]; const float* bm2 = (const float*)d_in[14];

  const int N = in_sizes[0] / 128;       // 50000 (multiple of 16)
  const int E = in_sizes[1] / 2;         // 800000
  const int G = out_size / 64;           // 500
  const int GP = ((G + 15) >> 4) << 4;   // 512 (padded rows for MLP GEMMs)
  const int* srcv = ei;
  const int* dstv = ei + E;

  float* ws   = (float*)d_ws;
  size_t off  = 0;
  float* dinv = ws + off; off += ((size_t)N + 255) & ~(size_t)255;
  float* Hcat = ws + off; off += (size_t)N * 288;   // f32 aggregation acc [h1|h2|h3]
  float* T    = ws + off; off += (size_t)N * 96;    // f32 GEMM temp / hjk
  float* emb  = ws + off; off += (size_t)GP * 96;
  float* t2   = ws + off; off += (size_t)GP * 96;
  unsigned short* xB    = (unsigned short*)(ws + off); off += (size_t)N * 64;   // x bf16 [N,128]
  unsigned short* HcatB = (unsigned short*)(ws + off); off += (size_t)N * 144;  // bf16 [N,288]
  unsigned short* embB  = (unsigned short*)(ws + off); off += (size_t)GP * 48;
  unsigned short* t2b   = (unsigned short*)(ws + off); off += (size_t)GP * 48;
  (void)ws_size; (void)n_in;

  auto cdiv = [](int a, int b) { return (a + b - 1) / b; };
  auto gblk = [&](int M) { return cdiv(cdiv(M, 16), 8); };

  // ---- degrees -> dinv ; zero accumulators ; x -> bf16 ----
  k_fill    <<<cdiv(N, 256), 256, 0, stream>>>(dinv, 1.0f, N);   // self-loop
  k_edge_deg<<<cdiv(E, 256), 256, 0, stream>>>(dstv, dinv, E);
  k_rsqrt   <<<cdiv(N, 256), 256, 0, stream>>>(dinv, N);
  k_zero4   <<<cdiv(N * 72, 256), 256, 0, stream>>>((float4*)Hcat, N * 72);
  k_cvt     <<<cdiv(N * 32, 256), 256, 0, stream>>>((const float4*)x, (uint2*)xB, N * 32);

  // ---- conv1: xB(128) -> HcatB[:,0:96] ----
  gemm_wmma<128, 6, 0><<<gblk(N), 256, 0, stream>>>(xB, 128, W0, nullptr, T, 96, N);
  k_edge_agg<<<cdiv(E * 32, 256), 256, 0, stream>>>(T, srcv, dstv, dinv, Hcat + 0, 288, E);
  k_finish  <<<cdiv(N * 24, 256), 256, 0, stream>>>(Hcat + 0, 288, T, dinv, b0, HcatB + 0, 288, N);

  // ---- conv2: HcatB[:,0:96] -> HcatB[:,96:192] ----
  gemm_wmma<96, 6, 0><<<gblk(N), 256, 0, stream>>>(HcatB + 0, 288, W1, nullptr, T, 96, N);
  k_edge_agg<<<cdiv(E * 32, 256), 256, 0, stream>>>(T, srcv, dstv, dinv, Hcat + 96, 288, E);
  k_finish  <<<cdiv(N * 24, 256), 256, 0, stream>>>(Hcat + 96, 288, T, dinv, b1, HcatB + 96, 288, N);

  // ---- conv3: HcatB[:,96:192] -> HcatB[:,192:288] ----
  gemm_wmma<96, 6, 0><<<gblk(N), 256, 0, stream>>>(HcatB + 96, 288, W2, nullptr, T, 96, N);
  k_edge_agg<<<cdiv(E * 32, 256), 256, 0, stream>>>(T, srcv, dstv, dinv, Hcat + 192, 288, E);
  k_finish  <<<cdiv(N * 24, 256), 256, 0, stream>>>(Hcat + 192, 288, T, dinv, b2, HcatB + 192, 288, N);

  // ---- JK 'cat' projection: HcatB[N,288] @ Wjk + bjk -> T[N,96] ----
  gemm_wmma<288, 6, 1><<<gblk(N), 256, 0, stream>>>(HcatB, 288, Wjk, bjk, T, 96, N);

  // ---- global add pool ----
  k_zero4<<<cdiv(GP * 24, 256), 256, 0, stream>>>((float4*)emb, GP * 24);
  k_pool <<<cdiv(N * 32, 256), 256, 0, stream>>>(T, bat, emb, N);
  k_cvt  <<<cdiv(GP * 24, 256), 256, 0, stream>>>((const float4*)emb, (uint2*)embB, GP * 24);

  // ---- fc_o MLP ----
  gemm_wmma<96, 6, 2><<<gblk(GP), 256, 0, stream>>>(embB, 96, Wm1, bm1, t2, 96, GP);
  k_cvt<<<cdiv(GP * 24, 256), 256, 0, stream>>>((const float4*)t2, (uint2*)t2b, GP * 24);
  gemm_wmma<96, 4, 1><<<gblk(G), 256, 0, stream>>>(t2b, 96, Wm2, bm2, (float*)d_out, 64, G);
}
